// CfdPoolGaussianSincosPos_49744311222707
// MI455X (gfx1250) — matrix operand order repaired and checked by reading, not compiled, not run on hardware
//
#include <hip/hip_runtime.h>
#include <cmath>

typedef __attribute__((ext_vector_type(16))) _Float16 v16h;
typedef __attribute__((ext_vector_type(8)))  _Float16 v8h;
typedef __attribute__((ext_vector_type(8)))  float    v8f;

#define N_NODES 131072
#define N_DST   16384

// Branch-free GELU (tanh form): trans-unit exp + rcp, co-executes with WMMA.
__device__ __forceinline__ float gelu_f(float x) {
    float y = 0.79788456080286536f * fmaf(0.044715f * x, x * x, x);
    float e = __expf(2.0f * y);                       // v_exp_f32
    float t = fmaf(-2.0f, __builtin_amdgcn_rcpf(e + 1.0f), 1.0f);  // tanh(y)
    return 0.5f * x * (1.0f + t);
}

__device__ __forceinline__ v8f wmma16(v16h a, v16h b, v8f c) {
    return __builtin_amdgcn_wmma_f32_16x16x32_f16(
        false, a, false, b, (short)0, c, false, false);
}

// A-fragment: 16x32 f16 tile row-major in LDS, leading dim ldk (halves).
// lanes 0-15: K=0..7 & 16..23 of row M=lane; lanes 16-31: K=8..15 & 24..31.
__device__ __forceinline__ v16h load_A_lds(const _Float16* tile, int ldk, int lane) {
    int m  = lane & 15;
    int hi = (lane >> 4) & 1;
    const _Float16* p = tile + m * ldk;
    v8h lo  = *(const v8h*)(p + hi * 8);
    v8h hi8 = *(const v8h*)(p + 16 + hi * 8);
    v16h a;
#pragma unroll
    for (int j = 0; j < 8; ++j) { a[j] = lo[j]; a[8 + j] = hi8[j]; }
    return a;
}

// B-fragment: 32x16 f16 (KxN), B[k][n] = W[n][k], W row-major (fan_out,fan_in).
// lanes 0-15: K=0..7 & 16..23 of col N=lane; lanes 16-31: K=8..15 & 24..31.
__device__ __forceinline__ v16h load_B_gl(const _Float16* W, int K, int n0, int k0, int lane) {
    int n  = lane & 15;
    int hi = (lane >> 4) & 1;
    const _Float16* p = W + (size_t)(n0 + n) * K + k0 + hi * 8;
    v8h b0 = *(const v8h*)p;
    v8h b1 = *(const v8h*)(p + 16);
    v16h b;
#pragma unroll
    for (int j = 0; j < 8; ++j) { b[j] = b0[j]; b[8 + j] = b1[j]; }
    return b;
}

// GELU+bias epilogue: write one 16x16 f32 accumulator tile as f16 to LDS.
__device__ __forceinline__ void store_gelu_tile(_Float16* dst, int ldn, v8f c,
                                                float bias, int hi, int col) {
#pragma unroll
    for (int r = 0; r < 8; ++r)
        dst[(hi * 8 + r) * ldn + col] = (_Float16)gelu_f(c[r] + bias);
}

// ---------------- weight f32 -> f16 conversion ----------------
__global__ __launch_bounds__(256) void cvt_f32_f16_kernel(const float* __restrict__ s,
                                                          _Float16* __restrict__ d, int n) {
    int i = blockIdx.x * 256 + threadIdx.x;
    if (i < n) d[i] = (_Float16)s[i];
}

// ---------------- node encoder: 64 nodes / block, 8 waves ----------------
__global__ __launch_bounds__(256) void node_encoder_kernel(
    const float* __restrict__ x, const float* __restrict__ pos,
    const float* __restrict__ w1, const float* __restrict__ b1,
    const _Float16* __restrict__ w2h, const float* __restrict__ b2,
    const _Float16* __restrict__ w3h, const float* __restrict__ b3,
    const float* __restrict__ bf, _Float16* __restrict__ h16)
{
    __shared__ _Float16 sh1[64 * 256];   // 32 KB
    __shared__ _Float16 sh2[64 * 256];   // 32 KB
    const int base = blockIdx.x * 64;
    const int t = threadIdx.x;

    // Layer 1: K=3 -> scalar FMA (WMMA would waste 29/32 of K).
    {
        float ww0 = w1[t * 3 + 0], ww1 = w1[t * 3 + 1], ww2 = w1[t * 3 + 2];
        float bb = b1[t];
        for (int i = 0; i < 64; ++i) {
            const float* xr = x + (size_t)(base + i) * 3;
            float v = fmaf(xr[0], ww0, fmaf(xr[1], ww1, fmaf(xr[2], ww2, bb)));
            sh1[i * 256 + t] = (_Float16)gelu_f(v);
        }
    }
    __syncthreads();

    const int wave = t >> 5, lane = t & 31;
    const int hi = (lane >> 4) & 1, col = lane & 15;

    // Layer 2: 64x256, K=256.  Register-blocked: 1 B fragment -> 4 WMMAs.
    for (int ni = 0; ni < 2; ++ni) {
        int nt = wave + ni * 8;
        v8f c[4] = {{}, {}, {}, {}};
#pragma unroll 4
        for (int k = 0; k < 256; k += 32) {
            v16h b = load_B_gl(w2h, 256, nt * 16, k, lane);
#pragma unroll
            for (int mt = 0; mt < 4; ++mt)
                c[mt] = wmma16(load_A_lds(sh1 + mt * 16 * 256 + k, 256, lane), b, c[mt]);
        }
        float bias = b2[nt * 16 + col];
#pragma unroll
        for (int mt = 0; mt < 4; ++mt)
            store_gelu_tile(sh2 + mt * 16 * 256 + nt * 16, 256, c[mt], bias, hi, col);
    }
    __syncthreads();

    // Layer 3 + fused Fourier positional add, store h as f16 to global.
    for (int ni = 0; ni < 2; ++ni) {
        int nt = wave + ni * 8;
        v8f c[4] = {{}, {}, {}, {}};
#pragma unroll 4
        for (int k = 0; k < 256; k += 32) {
            v16h b = load_B_gl(w3h, 256, nt * 16, k, lane);
#pragma unroll
            for (int mt = 0; mt < 4; ++mt)
                c[mt] = wmma16(load_A_lds(sh2 + mt * 16 * 256 + k, 256, lane), b, c[mt]);
        }
        int n = nt * 16 + col;
        int j = n & 127;
        float bf0 = bf[j * 2 + 0], bf1 = bf[j * 2 + 1];
        float bias = b3[n];
#pragma unroll
        for (int mt = 0; mt < 4; ++mt) {
#pragma unroll
            for (int r = 0; r < 8; ++r) {
                int m = base + mt * 16 + hi * 8 + r;
                float p0 = pos[m * 2 + 0], p1 = pos[m * 2 + 1];
                float proj = 6.28318530717958647f * fmaf(p0, bf0, p1 * bf1);
                float f = (n < 128) ? __cosf(proj) : __sinf(proj);
                h16[(size_t)m * 256 + n] = (_Float16)(c[mt][r] + bias + f);
            }
        }
    }
}

// ---------------- edge MLP + segment-mean: 4 dst (64 edges) / block ----------------
__global__ __launch_bounds__(256) void edge_mlp_pool_kernel(
    const long long* __restrict__ edges, const _Float16* __restrict__ h16,
    const _Float16* __restrict__ w1h, const float* __restrict__ b1,
    const _Float16* __restrict__ w2h, const float* __restrict__ b2,
    const _Float16* __restrict__ w3h, const float* __restrict__ b3,
    float* __restrict__ out)
{
    __shared__ _Float16 sA[64 * 512];    // 64 KB : concat[h_src,h_dst]; later layer2 out
    __shared__ _Float16 sB[64 * 512];    // 64 KB : layer1 out        (128 KB total)
    const int e0 = blockIdx.x * 64;
    const int t = threadIdx.x;

    // Gather: row e -> [h[src] | h[dst]], b128 loads (h is L2-resident, 67 MB f16).
    {
        int row = t >> 2, sub = t & 3;              // 4 threads per edge row
        long long dn = edges[(size_t)(e0 + row) * 2 + 0];
        long long sn = edges[(size_t)(e0 + row) * 2 + 1];
        const uint4* hs = (const uint4*)(h16 + (size_t)sn * 256);   // 32 x 16B
        const uint4* hd = (const uint4*)(h16 + (size_t)dn * 256);
        uint4* rp = (uint4*)(sA + row * 512);
#pragma unroll
        for (int i = sub; i < 32; i += 4) rp[i] = hs[i];
#pragma unroll
        for (int i = sub; i < 32; i += 4) rp[32 + i] = hd[i];
    }
    __syncthreads();

    const int wave = t >> 5, lane = t & 31;
    const int hi = (lane >> 4) & 1, col = lane & 15;

    // Layer 1: 64x512 = gelu(sA @ w1^T + b1), K=512.  1 B fragment -> 4 WMMAs.
    for (int ni = 0; ni < 4; ++ni) {
        int nt = wave + ni * 8;                      // 0..31
        v8f c[4] = {{}, {}, {}, {}};
#pragma unroll 4
        for (int k = 0; k < 512; k += 32) {
            v16h b = load_B_gl(w1h, 512, nt * 16, k, lane);
#pragma unroll
            for (int mt = 0; mt < 4; ++mt)
                c[mt] = wmma16(load_A_lds(sA + mt * 16 * 512 + k, 512, lane), b, c[mt]);
        }
        float bias = b1[nt * 16 + col];
#pragma unroll
        for (int mt = 0; mt < 4; ++mt)
            store_gelu_tile(sB + mt * 16 * 512 + nt * 16, 512, c[mt], bias, hi, col);
    }
    __syncthreads();

    // Layer 2: 64x256 = gelu(sB @ w2^T + b2), K=512; write into sA (stride 256).
    for (int ni = 0; ni < 2; ++ni) {
        int nt = wave + ni * 8;                      // 0..15
        v8f c[4] = {{}, {}, {}, {}};
#pragma unroll 4
        for (int k = 0; k < 512; k += 32) {
            v16h b = load_B_gl(w2h, 512, nt * 16, k, lane);
#pragma unroll
            for (int mt = 0; mt < 4; ++mt)
                c[mt] = wmma16(load_A_lds(sB + mt * 16 * 512 + k, 512, lane), b, c[mt]);
        }
        float bias = b2[nt * 16 + col];
#pragma unroll
        for (int mt = 0; mt < 4; ++mt)
            store_gelu_tile(sA + mt * 16 * 256 + nt * 16, 256, c[mt], bias, hi, col);
    }
    __syncthreads();

    // Layer 3 + pooled mean: each 16-row M-tile == the 16 edges of one dst
    // (dst = repeat(arange(N_DST),16)); segment-mean = column sum of the tile /16.
    for (int ni = 0; ni < 2; ++ni) {
        int nt = wave + ni * 8;
        v8f c[4] = {{}, {}, {}, {}};
#pragma unroll 4
        for (int k = 0; k < 256; k += 32) {
            v16h b = load_B_gl(w3h, 256, nt * 16, k, lane);
#pragma unroll
            for (int mt = 0; mt < 4; ++mt)
                c[mt] = wmma16(load_A_lds(sA + mt * 16 * 256 + k, 256, lane), b, c[mt]);
        }
        int n = nt * 16 + col;
        float bias = b3[n];
#pragma unroll
        for (int mt = 0; mt < 4; ++mt) {
            v8f cc = c[mt];
            float s = ((cc[0] + cc[1]) + (cc[2] + cc[3])) + ((cc[4] + cc[5]) + (cc[6] + cc[7]));
            s += __shfl_xor(s, 16, 32);      // add other 8 rows (wave32 halves)
            if (hi == 0) {
                int d = blockIdx.x * 4 + mt;
                out[(size_t)d * 256 + n] = s * (1.0f / 16.0f) + bias;
            }
        }
    }
}

extern "C" void kernel_launch(void* const* d_in, const int* in_sizes, int n_in,
                              void* d_out, int out_size, void* d_ws, size_t ws_size,
                              hipStream_t stream) {
    (void)in_sizes; (void)n_in; (void)out_size; (void)ws_size;
    const float*     x     = (const float*)d_in[0];
    const float*     pos   = (const float*)d_in[1];
    const long long* edges = (const long long*)d_in[2];
    const float* ip_w1 = (const float*)d_in[4];
    const float* ip_b1 = (const float*)d_in[5];
    const float* ip_w2 = (const float*)d_in[6];
    const float* ip_b2 = (const float*)d_in[7];
    const float* ip_w3 = (const float*)d_in[8];
    const float* ip_b3 = (const float*)d_in[9];
    const float* bf    = (const float*)d_in[10];
    const float* mw1   = (const float*)d_in[11];
    const float* mb1   = (const float*)d_in[12];
    const float* mw2   = (const float*)d_in[13];
    const float* mb2   = (const float*)d_in[14];
    const float* mw3   = (const float*)d_in[15];
    const float* mb3   = (const float*)d_in[16];

    _Float16* w2h  = (_Float16*)d_ws;
    _Float16* w3h  = w2h  + 256 * 256;
    _Float16* mw1h = w3h  + 256 * 256;
    _Float16* mw2h = mw1h + 512 * 512;
    _Float16* mw3h = mw2h + 256 * 512;
    _Float16* h16  = mw3h + 256 * 256;   // byte offset 1179648 (16B aligned)

    cvt_f32_f16_kernel<<<(256 * 256) / 256, 256, 0, stream>>>(ip_w2, w2h, 256 * 256);
    cvt_f32_f16_kernel<<<(256 * 256) / 256, 256, 0, stream>>>(ip_w3, w3h, 256 * 256);
    cvt_f32_f16_kernel<<<(512 * 512) / 256, 256, 0, stream>>>(mw1, mw1h, 512 * 512);
    cvt_f32_f16_kernel<<<(256 * 512) / 256, 256, 0, stream>>>(mw2, mw2h, 256 * 512);
    cvt_f32_f16_kernel<<<(256 * 256) / 256, 256, 0, stream>>>(mw3, mw3h, 256 * 256);

    node_encoder_kernel<<<N_NODES / 64, 256, 0, stream>>>(
        x, pos, ip_w1, ip_b1, w2h, ip_b2, w3h, ip_b3, bf, h16);

    edge_mlp_pool_kernel<<<N_DST / 4, 256, 0, stream>>>(
        edges, h16, mw1h, mb1, mw2h, mb2, mw3h, mb3, (float*)d_out);
}
